// myYOLOv3_45414984188348
// MI455X (gfx1250) — matrix operand order repaired and checked by reading, not compiled, not run on hardware
//
#include <hip/hip_runtime.h>
#include <hip/hip_bf16.h>

typedef _Float16 v16h __attribute__((ext_vector_type(16)));
typedef _Float16 v8h  __attribute__((ext_vector_type(8)));
typedef float    v8f  __attribute__((ext_vector_type(8)));

#define MODE_BN_LEAKY 0
#define MODE_PRED_BIAS 1

// ---------------------------------------------------------------------------
// Weight prep: W[Cout][Cin][taps] f32  ->  Wt[taps][Cout][Cin] f16
// (tap-major K ordering so the conv kernel's A/B fragments are contiguous)
// ---------------------------------------------------------------------------
__global__ void prep_weights_kernel(const float* __restrict__ W,
                                    _Float16* __restrict__ Wt,
                                    int Cout, int Cin, int taps, int total)
{
  for (int i = blockIdx.x * blockDim.x + threadIdx.x; i < total;
       i += gridDim.x * blockDim.x) {
    const int ci = i % Cin;
    const int m  = (i / Cin) % Cout;
    const int t  = i / (Cin * Cout);
    Wt[i] = (_Float16)W[((size_t)m * Cin + ci) * taps + t];
  }
}

// ---------------------------------------------------------------------------
// Implicit-GEMM conv, channels-last, via WMMA f32_16x16x32_f16.
// One wave computes 16 cout x (16*NT) voxels. Logical k = t*Cin + ci.
// Fragment maps (ISA 7.12.2, wave32):
//   A 16x32 f16 : lane l row m=l&15; half h -> k = gA + h (h<8) | gA+h+8,
//                 gA = (l>=16)?8:0  -> two contiguous 8-half runs (+0,+16)
//   B 32x16 f16 : lane l col n=l&15; half h -> k = gB + h (h<8) | gB+h+8, gB=(l>=16)?8:0
//                 -> two contiguous 8-half runs (+0,+16)
//   C/D f32     : lane l col n=l&15; vgpr v -> m = v + ((l>=16)?8:0)
//                 -> 8 consecutive m per lane (contiguous channels-last store)
// ---------------------------------------------------------------------------
template <int NT, int TAPS>
__global__ void __launch_bounds__(32)
conv_wmma_kernel(const _Float16* __restrict__ Wt,   // [TAPS][Cout][Cin] f16
                 const float* __restrict__ scale,   // [Cout] or null
                 const float* __restrict__ bias,    // [Cout]
                 const _Float16* __restrict__ X,    // [B][V][Cin] f16
                 _Float16* __restrict__ Yh,         // [B][V][Cout] f16 (or null)
                 float* __restrict__ Yf,            // [B][V][Cout] f32 (or null)
                 int Cin, int Cout, int S, int mode)
{
  const int V    = S * S * S;
  const int lane = threadIdx.x;
  const int b    = blockIdx.z;
  const int n0   = blockIdx.x * (16 * NT);
  const int m0   = blockIdx.y << 4;

  int  nj[NT], ax[NT], ay[NT], az[NT];
  bool nv[NT];
#pragma unroll
  for (int j = 0; j < NT; ++j) {
    nj[j] = n0 + j * 16 + (lane & 15);
    nv[j] = nj[j] < V;
    const int n = nv[j] ? nj[j] : 0;
    ax[j] = n / (S * S);
    ay[j] = (n / S) % S;
    az[j] = n % S;
  }

  const int groupA = (lane >> 4) << 3;    // 0 or 8
  const int groupB = (lane >> 4) << 3;    // 0 or 8
  const int mg     = m0 + (lane & 15);
  const bool mvalid = mg < Cout;

  const size_t planeW = (size_t)Cout * Cin;      // one tap of Wt
  const _Float16* wrow = Wt + (size_t)mg * Cin + groupA;
  const _Float16* xbat = X + (size_t)b * V * Cin;

  v8f acc[NT];
#pragma unroll
  for (int j = 0; j < NT; ++j)
    acc[j] = (v8f){0.f, 0.f, 0.f, 0.f, 0.f, 0.f, 0.f, 0.f};

  for (int t = 0; t < TAPS; ++t) {
    int dx = 0, dy = 0, dz = 0;
    if (TAPS == 27) { dx = t / 9 - 1; dy = (t / 3) % 3 - 1; dz = t % 3 - 1; }

    const _Float16* bptr[NT];
    bool ok[NT];
#pragma unroll
    for (int j = 0; j < NT; ++j) {
      const int sx = ax[j] + dx, sy = ay[j] + dy, sz = az[j] + dz;
      ok[j] = nv[j] & ((unsigned)sx < (unsigned)S) &
              ((unsigned)sy < (unsigned)S) & ((unsigned)sz < (unsigned)S);
      const int sv = ok[j] ? ((sx * S + sy) * S + sz) : 0;
      bptr[j] = xbat + (size_t)sv * Cin + groupB;
    }
    const _Float16* aptr = wrow + (size_t)t * planeW;

    for (int c0 = 0; c0 < Cin; c0 += 32) {
      // prefetch next K-step lines while the WMMA chain runs
      if (c0 + 32 < Cin) {
        __builtin_prefetch(aptr + c0 + 32, 0, 3);
#pragma unroll
        for (int j = 0; j < NT; ++j) __builtin_prefetch(bptr[j] + c0 + 32, 0, 3);
      }
      v16h A;
      if (mvalid) {
        const v8h a0 = *(const v8h*)(aptr + c0);
        const v8h a1 = *(const v8h*)(aptr + c0 + 16);
#pragma unroll
        for (int h = 0; h < 8; ++h) { A[h] = a0[h]; A[h + 8] = a1[h]; }
      } else {
#pragma unroll
        for (int h = 0; h < 16; ++h) A[h] = (_Float16)0.f;
      }
#pragma unroll
      for (int j = 0; j < NT; ++j) {
        v16h Bv;
        if (ok[j]) {
          const v8h b0 = *(const v8h*)(bptr[j] + c0);
          const v8h b1 = *(const v8h*)(bptr[j] + c0 + 16);
#pragma unroll
          for (int h = 0; h < 8; ++h) { Bv[h] = b0[h]; Bv[h + 8] = b1[h]; }
        } else {
#pragma unroll
          for (int h = 0; h < 16; ++h) Bv[h] = (_Float16)0.f;
        }
        acc[j] = __builtin_amdgcn_wmma_f32_16x16x32_f16(
            false, A, false, Bv, (short)0, acc[j], false, false);
      }
    }
  }

  const int mrow = m0 + ((lane >> 4) << 3);    // 8 consecutive output channels
#pragma unroll
  for (int j = 0; j < NT; ++j) {
    if (!nv[j]) continue;
    if (mode == MODE_BN_LEAKY) {
      v8h hv;
#pragma unroll
      for (int v = 0; v < 8; ++v) {
        float val = acc[j][v] * scale[mrow + v] + bias[mrow + v];
        val = (val > 0.f) ? val : 0.1f * val;
        hv[v] = (_Float16)val;
      }
      *(v8h*)(Yh + ((size_t)b * V + nj[j]) * Cout + mrow) = hv;
    } else {
#pragma unroll
      for (int v = 0; v < 8; ++v) {
        const int m = mrow + v;
        if (m < Cout)
          Yf[((size_t)b * V + nj[j]) * Cout + m] = acc[j][v] + bias[m];
      }
    }
  }
}

// f32 NCDHW -> f16 channels-last slice: dst[B][V][Ctot] at chOff
__global__ void cvt_concat_kernel(const float* __restrict__ src,
                                  _Float16* __restrict__ dst,
                                  int C, int V, int Ctot, int chOff, int total)
{
  for (int i = blockIdx.x * blockDim.x + threadIdx.x; i < total;
       i += gridDim.x * blockDim.x) {
    const int c = i % C;
    const int v = (i / C) % V;
    const int b = i / (C * V);
    dst[((size_t)b * V + v) * Ctot + chOff + c] =
        (_Float16)src[((size_t)b * C + c) * V + v];
  }
}

// trilinear 2x upsample, channels-last src[B][V][C] -> dst[B][VT][Ctot]+chOff
__global__ void upsample2x_kernel(const _Float16* __restrict__ src,
                                  _Float16* __restrict__ dst,
                                  int C, int S, int Ctot, int chOff, int total)
{
  const int T = 2 * S, V = S * S * S, VT = T * T * T;
  const float r = (float)(S - 1) / (float)(T - 1);
  for (int i = blockIdx.x * blockDim.x + threadIdx.x; i < total;
       i += gridDim.x * blockDim.x) {
    const int c = i % C;
    const int o = (i / C) % VT;
    const int b = i / (C * VT);
    const int oa = o / (T * T), ob = (o / T) % T, oc = o % T;
    const float pa = oa * r, pb = ob * r, pc = oc * r;
    const int ia = (int)pa, ib = (int)pb, ic = (int)pc;
    const float wa = pa - ia, wb = pb - ib, wc = pc - ic;
    const int ia1 = (ia + 1 < S) ? ia + 1 : S - 1;
    const int ib1 = (ib + 1 < S) ? ib + 1 : S - 1;
    const int ic1 = (ic + 1 < S) ? ic + 1 : S - 1;
    const _Float16* s0 = src + (size_t)b * V * C + c;
#define AT(x, y, z) ((float)s0[(size_t)(((x) * S + (y)) * S + (z)) * C])
    const float v00 = AT(ia, ib, ic) * (1.f - wc) + AT(ia, ib, ic1) * wc;
    const float v01 = AT(ia, ib1, ic) * (1.f - wc) + AT(ia, ib1, ic1) * wc;
    const float v10 = AT(ia1, ib, ic) * (1.f - wc) + AT(ia1, ib, ic1) * wc;
    const float v11 = AT(ia1, ib1, ic) * (1.f - wc) + AT(ia1, ib1, ic1) * wc;
#undef AT
    const float v0 = v00 * (1.f - wb) + v01 * wb;
    const float v1 = v10 * (1.f - wb) + v11 * wb;
    dst[((size_t)b * VT + o) * Ctot + chOff + c] =
        (_Float16)(v0 * (1.f - wa) + v1 * wa);
  }
}

__constant__ float d_anchors[3][3][3] = {
  {{10.f, 13.f, 12.f}, {16.f, 30.f, 20.f}, {33.f, 23.f, 28.f}},
  {{30.f, 61.f, 40.f}, {62.f, 45.f, 50.f}, {59.f, 119.f, 70.f}},
  {{116.f, 90.f, 100.f}, {156.f, 198.f, 150.f}, {373.f, 326.f, 300.f}}};

// decode one scale: P[B][V][81] (channels-last) -> rows [rowOff + s*3 + a]
__global__ void decode_kernel(const float* __restrict__ P,
                              float* __restrict__ boxes,
                              float* __restrict__ scores,
                              int S, int rowOff, int Ntot, float stride,
                              int scaleIdx, int total)
{
  const int V = S * S * S;
  for (int i = blockIdx.x * blockDim.x + threadIdx.x; i < total;
       i += gridDim.x * blockDim.x) {
    const int b = i / (V * 3);
    const int rem = i - b * V * 3;
    const int s = rem / 3;
    const int a = rem - s * 3;
    const float* pb = P + ((size_t)b * V + s) * 81 + a * 27;
    float p[27];
#pragma unroll
    for (int j = 0; j < 27; ++j) p[j] = pb[j];

    const float conf = 1.f / (1.f + expf(-p[0]));
    float mx = p[7];
#pragma unroll
    for (int j = 8; j < 27; ++j) mx = fmaxf(mx, p[j]);
    float e[20];
    float sum = 0.f;
#pragma unroll
    for (int j = 0; j < 20; ++j) { e[j] = expf(p[7 + j] - mx); sum += e[j]; }
    const float inv = 1.f / sum;

    const int gy = s / (S * S), gx = (s / S) % S, gz = s % S;
    const float cx = (1.f / (1.f + expf(-p[1])) + (float)gx) * stride;
    const float cy = (1.f / (1.f + expf(-p[2])) + (float)gy) * stride;
    const float cz = (1.f / (1.f + expf(-p[3])) + (float)gz) * stride;
    const float bw = expf(p[4]) * d_anchors[scaleIdx][a][0];
    const float bh = expf(p[5]) * d_anchors[scaleIdx][a][1];
    const float bd = expf(p[6]) * d_anchors[scaleIdx][a][2];

    const int row = rowOff + s * 3 + a;
    float* ob = boxes + ((size_t)b * Ntot + row) * 6;
    const float invs = 1.f / 160.f;
    ob[0] = fminf(fmaxf((cx - 0.5f * bw) * invs, 0.f), 1.f);
    ob[1] = fminf(fmaxf((cy - 0.5f * bh) * invs, 0.f), 1.f);
    ob[2] = fminf(fmaxf((cz - 0.5f * bd) * invs, 0.f), 1.f);
    ob[3] = fminf(fmaxf((cx + 0.5f * bw) * invs, 0.f), 1.f);
    ob[4] = fminf(fmaxf((cy + 0.5f * bh) * invs, 0.f), 1.f);
    ob[5] = fminf(fmaxf((cz + 0.5f * bd) * invs, 0.f), 1.f);
    float* os = scores + ((size_t)b * Ntot + row) * 20;
#pragma unroll
    for (int j = 0; j < 20; ++j) os[j] = conf * e[j] * inv;
  }
}

extern "C" void kernel_launch(void* const* d_in, const int* in_sizes, int n_in,
                              void* d_out, int out_size, void* d_ws, size_t ws_size,
                              hipStream_t stream)
{
  (void)in_sizes; (void)n_in; (void)out_size; (void)ws_size;
  const int B = 2;

  const float* fmp1 = (const float*)d_in[0];   // [2,256,20^3]
  const float* fmp2 = (const float*)d_in[1];   // [2,512,10^3]
  const float* fmp3 = (const float*)d_in[2];   // [2,1024,5^3]

  int idx = 3;
  auto next = [&]() { return (const float*)d_in[idx++]; };
  struct CP { const float *w, *s, *b; };
  auto nextCP = [&]() { CP p; p.w = next(); p.s = next(); p.b = next(); return p; };

  CP cs3[5]; for (int i = 0; i < 5; ++i) cs3[i] = nextCP();
  CP c13 = nextCP();
  CP ex3 = nextCP();
  const float* wp3 = next(); const float* bp3 = next();
  CP cs2[5]; for (int i = 0; i < 5; ++i) cs2[i] = nextCP();
  CP c12 = nextCP();
  CP ex2 = nextCP();
  const float* wp2 = next(); const float* bp2 = next();
  CP cs1[5]; for (int i = 0; i < 5; ++i) cs1[i] = nextCP();
  CP ex1 = nextCP();
  const float* wp1 = next(); const float* bp1 = next();

  // workspace layout (~57 MB)
  uint8_t* ws = (uint8_t*)d_ws;
  size_t off = 0;
  auto alloc = [&](size_t bytes) -> void* {
    void* p = ws + off;
    off = (off + bytes + 255) & ~(size_t)255;
    return p;
  };
  _Float16* WT = (_Float16*)alloc((size_t)27 * 1024 * 512 * 2);  // weight scratch
  _Float16* X0 = (_Float16*)alloc((size_t)B * 1024 * 125 * 2);
  _Float16* X1 = (_Float16*)alloc((size_t)B * 512 * 125 * 2);
  _Float16* X2 = (_Float16*)alloc((size_t)B * 256 * 125 * 2);
  float*    P3 = (float*)   alloc((size_t)B * 81 * 125 * 4);
  _Float16* C2 = (_Float16*)alloc((size_t)B * 768 * 1000 * 2);
  _Float16* Y1 = (_Float16*)alloc((size_t)B * 256 * 1000 * 2);
  _Float16* Y2 = (_Float16*)alloc((size_t)B * 128 * 1000 * 2);
  float*    P2 = (float*)   alloc((size_t)B * 81 * 1000 * 4);
  _Float16* C1 = (_Float16*)alloc((size_t)B * 384 * 8000 * 2);
  _Float16* Z1 = (_Float16*)alloc((size_t)B * 128 * 8000 * 2);
  float*    P1 = (float*)   alloc((size_t)B * 81 * 8000 * 4);

  auto run = [&](const float* W, const float* s, const float* b,
                 const _Float16* X, _Float16* Yh, float* Yf,
                 int Cin, int Cout, int S, int ks, int mode) {
    const int taps = (ks == 3) ? 27 : 1;
    const int wtot = taps * Cout * Cin;
    prep_weights_kernel<<<(wtot + 255) / 256, 256, 0, stream>>>(
        W, WT, Cout, Cin, taps, wtot);
    const int V = S * S * S;
    dim3 g((V + 63) / 64, (Cout + 15) / 16, B);
    if (taps == 27)
      conv_wmma_kernel<4, 27><<<g, dim3(32, 1, 1), 0, stream>>>(
          WT, s, b, X, Yh, Yf, Cin, Cout, S, mode);
    else
      conv_wmma_kernel<4, 1><<<g, dim3(32, 1, 1), 0, stream>>>(
          WT, s, b, X, Yh, Yf, Cin, Cout, S, mode);
  };

  // ----- stage 3 (5^3) -----
  { int t = B * 1024 * 125;
    cvt_concat_kernel<<<(t + 255) / 256, 256, 0, stream>>>(fmp3, X0, 1024, 125, 1024, 0, t); }
  run(cs3[0].w, cs3[0].s, cs3[0].b, X0, X1, nullptr, 1024, 512, 5, 1, MODE_BN_LEAKY);
  run(cs3[1].w, cs3[1].s, cs3[1].b, X1, X0, nullptr, 512, 1024, 5, 3, MODE_BN_LEAKY);
  run(cs3[2].w, cs3[2].s, cs3[2].b, X0, X1, nullptr, 1024, 512, 5, 1, MODE_BN_LEAKY);
  run(cs3[3].w, cs3[3].s, cs3[3].b, X1, X0, nullptr, 512, 1024, 5, 3, MODE_BN_LEAKY);
  run(cs3[4].w, cs3[4].s, cs3[4].b, X0, X1, nullptr, 1024, 512, 5, 1, MODE_BN_LEAKY); // X1 = f3
  run(ex3.w, ex3.s, ex3.b, X1, X0, nullptr, 512, 1024, 5, 3, MODE_BN_LEAKY);
  run(wp3, nullptr, bp3, X0, nullptr, P3, 1024, 81, 5, 1, MODE_PRED_BIAS);
  run(c13.w, c13.s, c13.b, X1, X2, nullptr, 512, 256, 5, 1, MODE_BN_LEAKY);
  { int t = B * 256 * 1000;
    upsample2x_kernel<<<(t + 255) / 256, 256, 0, stream>>>(X2, C2, 256, 5, 768, 512, t); }
  { int t = B * 512 * 1000;
    cvt_concat_kernel<<<(t + 255) / 256, 256, 0, stream>>>(fmp2, C2, 512, 1000, 768, 0, t); }

  // ----- stage 2 (10^3) -----
  run(cs2[0].w, cs2[0].s, cs2[0].b, C2, Y1, nullptr, 768, 256, 10, 1, MODE_BN_LEAKY);
  run(cs2[1].w, cs2[1].s, cs2[1].b, Y1, C2, nullptr, 256, 512, 10, 3, MODE_BN_LEAKY);
  run(cs2[2].w, cs2[2].s, cs2[2].b, C2, Y1, nullptr, 512, 256, 10, 1, MODE_BN_LEAKY);
  run(cs2[3].w, cs2[3].s, cs2[3].b, Y1, C2, nullptr, 256, 512, 10, 3, MODE_BN_LEAKY);
  run(cs2[4].w, cs2[4].s, cs2[4].b, C2, Y1, nullptr, 512, 256, 10, 1, MODE_BN_LEAKY); // Y1 = f2
  run(ex2.w, ex2.s, ex2.b, Y1, C2, nullptr, 256, 512, 10, 3, MODE_BN_LEAKY);
  run(wp2, nullptr, bp2, C2, nullptr, P2, 512, 81, 10, 1, MODE_PRED_BIAS);
  run(c12.w, c12.s, c12.b, Y1, Y2, nullptr, 256, 128, 10, 1, MODE_BN_LEAKY);
  { int t = B * 128 * 8000;
    upsample2x_kernel<<<(t + 255) / 256, 256, 0, stream>>>(Y2, C1, 128, 10, 384, 256, t); }
  { int t = B * 256 * 8000;
    cvt_concat_kernel<<<(t + 255) / 256, 256, 0, stream>>>(fmp1, C1, 256, 8000, 384, 0, t); }

  // ----- stage 1 (20^3) -----
  run(cs1[0].w, cs1[0].s, cs1[0].b, C1, Z1, nullptr, 384, 128, 20, 1, MODE_BN_LEAKY);
  run(cs1[1].w, cs1[1].s, cs1[1].b, Z1, C1, nullptr, 128, 256, 20, 3, MODE_BN_LEAKY);
  run(cs1[2].w, cs1[2].s, cs1[2].b, C1, Z1, nullptr, 256, 128, 20, 1, MODE_BN_LEAKY);
  run(cs1[3].w, cs1[3].s, cs1[3].b, Z1, C1, nullptr, 128, 256, 20, 3, MODE_BN_LEAKY);
  run(cs1[4].w, cs1[4].s, cs1[4].b, C1, Z1, nullptr, 256, 128, 20, 1, MODE_BN_LEAKY); // Z1 = f1
  run(ex1.w, ex1.s, ex1.b, Z1, C1, nullptr, 128, 256, 20, 3, MODE_BN_LEAKY);
  run(wp1, nullptr, bp1, C1, nullptr, P1, 256, 81, 20, 1, MODE_PRED_BIAS);

  // ----- decode -----
  const int Ntot = 3 * (8000 + 1000 + 125);   // 27375
  float* boxes  = (float*)d_out;
  float* scores = boxes + (size_t)B * Ntot * 6;
  { int t = B * 8000 * 3;
    decode_kernel<<<(t + 255) / 256, 256, 0, stream>>>(P1, boxes, scores, 20, 0,     Ntot, 8.f,  0, t); }
  { int t = B * 1000 * 3;
    decode_kernel<<<(t + 255) / 256, 256, 0, stream>>>(P2, boxes, scores, 10, 24000, Ntot, 16.f, 1, t); }
  { int t = B * 125 * 3;
    decode_kernel<<<(t + 255) / 256, 256, 0, stream>>>(P3, boxes, scores, 5,  27000, Ntot, 32.f, 2, t); }
}
